// PatchedSelfAttention_90082644066542
// MI455X (gfx1250) — compile-verified, matrix operand order checked by reading, not run on hardware
//
#include <hip/hip_runtime.h>
#include <hip/hip_bf16.h>
#include <cstdint>
#include <cstddef>

// ---------------------------------------------------------------------------
// Problem constants (fixed by the reference)
// ---------------------------------------------------------------------------
#define B_       4
#define LQ_      512
#define NH_      32
#define NKV_     2
#define HD_      128
#define HIDDEN_  4096
#define QKV_OUT_ 4608          // NH*HD + 2*NKV*HD
#define T_       2048          // B * LQ
#define BS_      64
#define MAXB_    24
#define MAXKV_   1536          // MAXB * BS
#define KVSTRIDE_ (NKV_*HD_)   // 256

typedef __bf16 bf16_t;
typedef __attribute__((ext_vector_type(16))) bf16_t v16bf;
typedef __attribute__((ext_vector_type(8)))  bf16_t v8bf;
typedef __attribute__((ext_vector_type(4)))  bf16_t v4bf;
typedef __attribute__((ext_vector_type(8)))  float  v8f;
typedef __attribute__((ext_vector_type(4)))  float  v4f;

// ---------------------------------------------------------------------------
// CDNA5 helpers
// ---------------------------------------------------------------------------
__device__ __forceinline__ v8f wmma_bf16(v16bf a, v16bf b, v8f c) {
  return __builtin_amdgcn_wmma_f32_16x16x32_bf16(
      /*neg_a=*/false, a, /*neg_b=*/false, b,
      /*c_mod=*/(short)0, c, /*reuse_a=*/false, /*reuse_b=*/false);
}

// A/B fragment: element j at k = ((j&8)?16:0) + half*8 + (j&7)
// -> two contiguous 16-byte LDS chunks at [base] and [base+16].
__device__ __forceinline__ v16bf load_frag(const bf16_t* p) {
  v8bf lo = *(const v8bf*)(p);
  v8bf hi = *(const v8bf*)(p + 16);
  return __builtin_shufflevector(lo, hi, 0,1,2,3,4,5,6,7,8,9,10,11,12,13,14,15);
}

// Async global->LDS 16-byte copy (ASYNCcnt tracked).
// LDS destination = low 32 bits of the generic shared pointer (LDS aperture
// truncates addr[31:0] to the LDS byte address).
__device__ __forceinline__ void async_copy16(uint32_t lds_addr, uint64_t gaddr) {
  asm volatile("global_load_async_to_lds_b128 %0, %1, off"
               :: "v"(lds_addr), "v"(gaddr) : "memory");
}
__device__ __forceinline__ void wait_async_le4() {
  asm volatile("s_wait_asynccnt 0x4" ::: "memory");
}
__device__ __forceinline__ void wait_async_0() {
  asm volatile("s_wait_asynccnt 0x0" ::: "memory");
}
__device__ __forceinline__ uint32_t lds_addr_of(const void* p) {
  return (uint32_t)(size_t)p;
}

__device__ __forceinline__ float redmax16(float v) {
#pragma unroll
  for (int m = 1; m <= 8; m <<= 1) v = fmaxf(v, __shfl_xor(v, m, 32));
  return v;
}
__device__ __forceinline__ float redsum16(float v) {
#pragma unroll
  for (int m = 1; m <= 8; m <<= 1) v += __shfl_xor(v, m, 32);
  return v;
}

// ---------------------------------------------------------------------------
// Kernel: f32 -> bf16 bulk convert (8 elems/thread)
// ---------------------------------------------------------------------------
__global__ __launch_bounds__(256)
void f32_to_bf16(const float* __restrict__ s, bf16_t* __restrict__ d, int n8)
{
  const int i = blockIdx.x * 256 + threadIdx.x;
  if (i >= n8) return;
  const v4f f0 = *(const v4f*)(s + (size_t)i * 8);
  const v4f f1 = *(const v4f*)(s + (size_t)i * 8 + 4);
  v8bf o;
#pragma unroll
  for (int j = 0; j < 4; ++j) { o[j] = (bf16_t)f0[j]; o[j + 4] = (bf16_t)f1[j]; }
  *(v8bf*)(d + (size_t)i * 8) = o;
}

// ---------------------------------------------------------------------------
// Kernel: C[M,N] = A[M,K] @ W[N,K]^T + bias
// A, W bf16; C f32. 256 threads = 8 waves; 128x128 tile; double-buffered LDS
// filled by GLOBAL_LOAD_ASYNC_TO_LDS_B128; batched fragment loads so the 8
// WMMAs issue back-to-back behind a single s_wait_dscnt.
// ---------------------------------------------------------------------------
__global__ __launch_bounds__(256)
void gemm_bf16_async(const bf16_t* __restrict__ A, const bf16_t* __restrict__ W,
                     const float* __restrict__ bias, float* __restrict__ C,
                     int M, int N, int K)
{
  (void)M;
  __shared__ bf16_t Al[2][128][40];   // stride 40 => 80B rows, 16B chunks
  __shared__ bf16_t Bl[2][128][40];

  const int m0   = blockIdx.y * 128;
  const int n0   = blockIdx.x * 128;
  const int tid  = threadIdx.x;
  const int wave = tid >> 5;
  const int lane = tid & 31;
  const int half = lane >> 4;
  const int r    = lane & 15;

  v8f acc[8];
  const v8f vzero = {0.f,0.f,0.f,0.f,0.f,0.f,0.f,0.f};
#pragma unroll
  for (int t = 0; t < 8; ++t) acc[t] = vzero;

  // staging map: thread (srow, sseg) copies 16 bf16 (=32B, two b128 chunks)
  const int srow = tid >> 1;   // 0..127
  const int sseg = tid & 1;    // k sub-range: sseg*16 .. +15
  const bf16_t* arow_g = A + (size_t)(m0 + srow) * K + sseg * 16;
  const bf16_t* wrow_g = W + (size_t)(n0 + srow) * K + sseg * 16;

#define GEMM_ISSUE(bufi, kk0)                                                  \
  {                                                                            \
    const uint64_t ga = (uint64_t)(arow_g + (kk0));                            \
    const uint64_t gb = (uint64_t)(wrow_g + (kk0));                            \
    const uint32_t la = lds_addr_of(&Al[(bufi)][srow][sseg * 16]);             \
    const uint32_t lb = lds_addr_of(&Bl[(bufi)][srow][sseg * 16]);             \
    async_copy16(la,      ga);                                                 \
    async_copy16(la + 16, ga + 16);                                            \
    async_copy16(lb,      gb);                                                 \
    async_copy16(lb + 16, gb + 16);                                            \
  }

  GEMM_ISSUE(0, 0);   // prologue: 4 async ops outstanding

  for (int k0 = 0; k0 < K; k0 += 32) {
    const int buf = (k0 >> 5) & 1;
    if (k0 + 32 < K) {
      GEMM_ISSUE(buf ^ 1, k0 + 32);   // 8 outstanding
      wait_async_le4();               // oldest 4 (this buf) complete, in order
    } else {
      wait_async_0();
    }
    __syncthreads();

    // batch all fragment loads, then stream 8 WMMAs back-to-back
    v16bf afrag = load_frag(&Al[buf][wave * 16 + r][half * 8]);
    v16bf bfr[8];
#pragma unroll
    for (int nt = 0; nt < 8; ++nt)
      bfr[nt] = load_frag(&Bl[buf][nt * 16 + r][half * 8]);
#pragma unroll
    for (int nt = 0; nt < 8; ++nt)
      acc[nt] = wmma_bf16(afrag, bfr[nt], acc[nt]);
    __syncthreads();   // all reads of `buf` done before next issue overwrites it
  }
#undef GEMM_ISSUE

  const bool hb = (bias != nullptr);
#pragma unroll
  for (int nt = 0; nt < 8; ++nt) {
    const int col = n0 + nt * 16 + r;
    const float bv = hb ? bias[col] : 0.0f;
#pragma unroll
    for (int i = 0; i < 8; ++i) {
      const int row = m0 + wave * 16 + i + half * 8;   // WMMA C layout
      C[(size_t)row * N + col] = acc[nt][i] + bv;
    }
  }
}

// ---------------------------------------------------------------------------
// Kernel: RoPE on Q/K + scatter new K/V into contiguous bf16 KV buffers
// qkv [T,4608] f32 ; rope [T,32,2] f32 ; qb [T,32,128] bf16 ; Kg/Vg [B,1536,2,128]
// ---------------------------------------------------------------------------
__global__ __launch_bounds__(256)
void rope_scatter(const float* __restrict__ qkv, const float* __restrict__ rope,
                  const int* __restrict__ hist,
                  bf16_t* __restrict__ qb, bf16_t* __restrict__ Kg,
                  bf16_t* __restrict__ Vg)
{
  const int t = blockIdx.x;
  const int b = t >> 9;
  const int q = t & (LQ_ - 1);
  const int kvpos = hist[b] + q;
  const float* row = qkv + (size_t)t * QKV_OUT_;
  const float* rp  = rope + (size_t)t * 64;

  for (int idx = threadIdx.x; idx < (NH_ + NKV_) * 64; idx += 256) {
    const int head = idx >> 6;
    const int j    = idx & 63;
    const bool is_k = head >= NH_;
    const int hh   = is_k ? (head - NH_) : head;
    const int base = is_k ? (NH_ * HD_ + hh * HD_) : hh * HD_;
    const int d0   = 2 * j;                  // rotary (j<32) / passthrough (j>=32)
    const float x0 = row[base + d0];
    const float x1 = row[base + d0 + 1];
    float o0, o1;
    if (j < 32) {
      const float c = rp[2 * j], s = rp[2 * j + 1];
      o0 = x0 * c - x1 * s;
      o1 = x1 * c + x0 * s;
    } else { o0 = x0; o1 = x1; }
    if (!is_k) {
      bf16_t* dq = qb + ((size_t)t * NH_ + hh) * HD_ + d0;
      dq[0] = (bf16_t)o0; dq[1] = (bf16_t)o1;
    } else {
      bf16_t* dk = Kg + (((size_t)b * MAXKV_ + kvpos) * NKV_ + hh) * HD_ + d0;
      dk[0] = (bf16_t)o0; dk[1] = (bf16_t)o1;
    }
  }
  for (int idx = threadIdx.x; idx < NKV_ * HD_; idx += 256) {
    const int hh = idx >> 7, d = idx & 127;
    const float v = row[NH_ * HD_ + NKV_ * HD_ + hh * HD_ + d];
    Vg[(((size_t)b * MAXKV_ + kvpos) * NKV_ + hh) * HD_ + d] = (bf16_t)v;
  }
}

// ---------------------------------------------------------------------------
// Kernel: gather paged-cache history into contiguous bf16 KV buffers
// ---------------------------------------------------------------------------
__global__ __launch_bounds__(256)
void gather_hist(const float* __restrict__ ck, const float* __restrict__ cv,
                 const int* __restrict__ boff, const int* __restrict__ hist,
                 bf16_t* __restrict__ Kg, bf16_t* __restrict__ Vg)
{
  const int g   = blockIdx.x * 256 + threadIdx.x;
  const int b   = g >> 18;
  const int rem = g & ((1 << 18) - 1);
  const int p   = rem >> 8;                         // kv position < 1024
  const int c   = rem & 255;                        // kvh*128 + d
  if (p >= hist[b]) return;
  const int blk = boff[b * MAXB_ + (p >> 6)];
  const size_t src = ((size_t)blk * BS_ + (p & (BS_ - 1))) * KVSTRIDE_ + c;
  const size_t dst = ((size_t)b * MAXKV_ + p) * KVSTRIDE_ + c;
  Kg[dst] = (bf16_t)ck[src];
  Vg[dst] = (bf16_t)cv[src];
}

// ---------------------------------------------------------------------------
// Kernel: causal GQA flash attention, bf16 WMMA + f32 online softmax
// grid (LQ/64, NH, B), block 128 (4 waves, 16 Q rows each). bf16 output.
// K chunk staged with async copies; V chunk transposed through VGPRs.
// Unmasked garbage beyond kv_end is safe: those columns are always causally
// masked (P -> 0) and clamped addresses stay inside the KV buffers.
// ---------------------------------------------------------------------------
__global__ __launch_bounds__(128)
void flash_attn(const bf16_t* __restrict__ qb, const bf16_t* __restrict__ Kg,
                const bf16_t* __restrict__ Vg, const int* __restrict__ hist,
                bf16_t* __restrict__ attn)
{
  __shared__ bf16_t Kl[32][136];      // [kv][d] chunk (272B rows, 16B chunks)
  __shared__ bf16_t Vlt[128][40];     // [d][kv] transposed V chunk
  __shared__ bf16_t Pl[4][16][40];    // per-wave P transpose scratch

  const int qt = blockIdx.x, h = blockIdx.y, b = blockIdx.z;
  const int kvh  = h >> 4;            // 16 Q heads per KV head
  const int wave = threadIdx.x >> 5;
  const int lane = threadIdx.x & 31;
  const int half = lane >> 4;
  const int r    = lane & 15;
  const int hist_b = hist[b];
  const int q0b = qt * 64;
  const int q0w = q0b + wave * 16;
  const int kv_end = hist_b + q0b + 64;     // block-uniform causal bound
  const float scale = 0.08838834764831845f; // 1/sqrt(128)

  // Q A-fragments (row = q0w + r), K dim split into 4 chunks of 32
  const bf16_t* qrow = qb + (((size_t)(b * LQ_ + q0w + r)) * NH_ + h) * HD_;
  v16bf qa[4];
#pragma unroll
  for (int kk = 0; kk < 4; ++kk) qa[kk] = load_frag(qrow + kk * 32 + half * 8);

  const v8f vzero = {0.f,0.f,0.f,0.f,0.f,0.f,0.f,0.f};
  v8f O[8];
#pragma unroll
  for (int t = 0; t < 8; ++t) O[t] = vzero;
  float mrow[8], lrow[8];
#pragma unroll
  for (int i = 0; i < 8; ++i) { mrow[i] = -3.0e38f; lrow[i] = 0.0f; }

  const bf16_t* Kbase = Kg + ((size_t)b * MAXKV_ * NKV_ + kvh) * HD_;
  const bf16_t* Vbase = Vg + ((size_t)b * MAXKV_ * NKV_ + kvh) * HD_;
  // K staging map: 32 rows x 128d; thread -> (row, 32d) = 4 async b128
  const int ksrow = threadIdx.x >> 2;
  const int ksseg = threadIdx.x & 3;
  // V staging map: 4 kv rows x 8 d per thread, packed b64 stores
  const int vgrp = threadIdx.x >> 4;   // 0..7 -> kv rows vgrp*4..+3
  const int vdc  = threadIdx.x & 15;   // d chunk vdc*8..+7

  for (int kc = 0; kc < kv_end; kc += 32) {
    // ---- async-stage K chunk ----
    {
      int krow = kc + ksrow; if (krow >= MAXKV_) krow = MAXKV_ - 1;
      const uint64_t g = (uint64_t)(Kbase + (size_t)krow * KVSTRIDE_ + ksseg * 32);
      const uint32_t l = lds_addr_of(&Kl[ksrow][ksseg * 32]);
#pragma unroll
      for (int c = 0; c < 4; ++c) async_copy16(l + c * 16, g + c * 16);
    }
    // ---- stage V chunk transposed (overlaps with async K) ----
    {
      v8bf vv[4];
#pragma unroll
      for (int rr = 0; rr < 4; ++rr) {
        int vrow = kc + vgrp * 4 + rr; if (vrow >= MAXKV_) vrow = MAXKV_ - 1;
        vv[rr] = *(const v8bf*)(Vbase + (size_t)vrow * KVSTRIDE_ + vdc * 8);
      }
#pragma unroll
      for (int dd = 0; dd < 8; ++dd) {
        v4bf p = {vv[0][dd], vv[1][dd], vv[2][dd], vv[3][dd]};
        *(v4bf*)&Vlt[vdc * 8 + dd][vgrp * 4] = p;
      }
    }
    // prefetch two chunks ahead
    if (kc + 64 < kv_end) {
      __builtin_prefetch(Kbase + (size_t)(kc + 64 + ksrow) * KVSTRIDE_ + ksseg * 32, 0, 1);
      __builtin_prefetch(Vbase + (size_t)(kc + 64 + ksrow) * KVSTRIDE_ + ksseg * 32, 0, 1);
    }
    wait_async_0();
    __syncthreads();

    // ---- S = Q @ K^T : preload 8 K fragments, then 8 WMMAs back-to-back ----
    v16bf kb[8];
#pragma unroll
    for (int kk = 0; kk < 4; ++kk) {
      kb[2 * kk]     = load_frag(&Kl[r][kk * 32 + half * 8]);      // cols kc..+15
      kb[2 * kk + 1] = load_frag(&Kl[16 + r][kk * 32 + half * 8]); // cols kc+16..+31
    }
    v8f s0 = vzero, s1 = vzero;
#pragma unroll
    for (int kk = 0; kk < 4; ++kk) {
      s0 = wmma_bf16(qa[kk], kb[2 * kk],     s0);
      s1 = wmma_bf16(qa[kk], kb[2 * kk + 1], s1);
    }

    // ---- scale, causal mask, online softmax (C layout: row=i+8*half) ----
    const int qpb = hist_b + q0w + half * 8;
#pragma unroll
    for (int i = 0; i < 8; ++i) {
      float v0 = s0[i] * scale;
      float v1 = s1[i] * scale;
      if (kc + r      > qpb + i) v0 = -1.0e30f;
      if (kc + 16 + r > qpb + i) v1 = -1.0e30f;
      const float cm = redmax16(fmaxf(v0, v1));
      const float mn = fmaxf(mrow[i], cm);
      const float alpha = __expf(mrow[i] - mn);
      const float p0 = __expf(v0 - mn);
      const float p1 = __expf(v1 - mn);
      const float rs = redsum16(p0 + p1);
      lrow[i] = lrow[i] * alpha + rs;
      mrow[i] = mn;
#pragma unroll
      for (int t = 0; t < 8; ++t) O[t][i] *= alpha;
      Pl[wave][i + half * 8][r]      = (bf16_t)p0;
      Pl[wave][i + half * 8][16 + r] = (bf16_t)p1;
    }
    __builtin_amdgcn_wave_barrier();

    // ---- O += P @ V : preload 8 V fragments, then 8 WMMAs back-to-back ----
    v16bf pa = load_frag(&Pl[wave][r][half * 8]);
    v16bf vbf[8];
#pragma unroll
    for (int dt = 0; dt < 8; ++dt)
      vbf[dt] = load_frag(&Vlt[dt * 16 + r][half * 8]);
#pragma unroll
    for (int dt = 0; dt < 8; ++dt)
      O[dt] = wmma_bf16(pa, vbf[dt], O[dt]);
    __syncthreads();
  }

  // ---- normalize and write bf16 attention output [T, 4096] ----
#pragma unroll
  for (int i = 0; i < 8; ++i) {
    const float inv = 1.0f / lrow[i];
    const int t = b * LQ_ + q0w + i + half * 8;
    bf16_t* orow = attn + (size_t)t * HIDDEN_ + h * HD_;
#pragma unroll
    for (int dt = 0; dt < 8; ++dt) orow[dt * 16 + r] = (bf16_t)(O[dt][i] * inv);
  }
}

// ---------------------------------------------------------------------------
// Launcher
// ---------------------------------------------------------------------------
extern "C" void kernel_launch(void* const* d_in, const int* in_sizes, int n_in,
                              void* d_out, int out_size, void* d_ws, size_t ws_size,
                              hipStream_t stream)
{
  (void)in_sizes; (void)n_in; (void)out_size; (void)ws_size;

  const float* hidden  = (const float*)d_in[0];
  const float* rope    = (const float*)d_in[1];
  const float* qkv_w   = (const float*)d_in[2];
  const float* qkv_b   = (const float*)d_in[3];
  const float* dense_w = (const float*)d_in[4];
  const float* cache_k = (const float*)d_in[5];
  const float* cache_v = (const float*)d_in[6];
  const int*   boff    = (const int*)d_in[7];
  const int*   hist    = (const int*)d_in[8];
  float* out = (float*)d_out;

  char* ws = (char*)d_ws;
  size_t off = 0;
  float*  qkv    = (float*)(ws + off);  off += (size_t)T_ * QKV_OUT_ * sizeof(float);
  bf16_t* qbuf   = (bf16_t*)(ws + off); off += (size_t)T_ * HIDDEN_ * sizeof(bf16_t);
  bf16_t* Kg     = (bf16_t*)(ws + off); off += (size_t)B_ * MAXKV_ * KVSTRIDE_ * sizeof(bf16_t);
  bf16_t* Vg     = (bf16_t*)(ws + off); off += (size_t)B_ * MAXKV_ * KVSTRIDE_ * sizeof(bf16_t);
  bf16_t* attnb  = (bf16_t*)(ws + off); off += (size_t)T_ * HIDDEN_ * sizeof(bf16_t);
  bf16_t* hid_bf = (bf16_t*)(ws + off); off += (size_t)T_ * HIDDEN_ * sizeof(bf16_t);
  bf16_t* qw_bf  = (bf16_t*)(ws + off); off += (size_t)QKV_OUT_ * HIDDEN_ * sizeof(bf16_t);
  bf16_t* dw_bf  = (bf16_t*)(ws + off); off += (size_t)HIDDEN_ * HIDDEN_ * sizeof(bf16_t);

  // 0) bf16 prepass (weights land in L2-resident bf16 form)
  f32_to_bf16<<<(T_ * HIDDEN_ / 8 + 255) / 256, 256, 0, stream>>>(hidden, hid_bf, T_ * HIDDEN_ / 8);
  f32_to_bf16<<<(QKV_OUT_ * HIDDEN_ / 8 + 255) / 256, 256, 0, stream>>>(qkv_w, qw_bf, QKV_OUT_ * HIDDEN_ / 8);
  f32_to_bf16<<<(HIDDEN_ * HIDDEN_ / 8 + 255) / 256, 256, 0, stream>>>(dense_w, dw_bf, HIDDEN_ * HIDDEN_ / 8);

  // 1) QKV projection + bias (WMMA, async-LDS double-buffered)
  gemm_bf16_async<<<dim3(QKV_OUT_ / 128, T_ / 128), 256, 0, stream>>>(
      hid_bf, qw_bf, qkv_b, qkv, T_, QKV_OUT_, HIDDEN_);
  // 2) RoPE + scatter fresh K/V
  rope_scatter<<<T_, 256, 0, stream>>>(qkv, rope, hist, qbuf, Kg, Vg);
  // 3) gather paged history into contiguous KV
  gather_hist<<<B_ * 1024, 256, 0, stream>>>(cache_k, cache_v, boff, hist, Kg, Vg);
  // 4) flash attention (WMMA), bf16 output
  flash_attn<<<dim3(LQ_ / 64, NH_, B_), 128, 0, stream>>>(qbuf, Kg, Vg, hist, attnb);
  // 5) output projection (WMMA, async-LDS double-buffered)
  gemm_bf16_async<<<dim3(HIDDEN_ / 128, T_ / 128), 256, 0, stream>>>(
      attnb, dw_bf, nullptr, out, T_, HIDDEN_, HIDDEN_);
}
